// GPT_56126632624712
// MI455X (gfx1250) — matrix operand order repaired and checked by reading, not compiled, not run on hardware
//
#include <hip/hip_runtime.h>

#define EPSF 1.1920929e-07f

typedef __attribute__((ext_vector_type(16))) __bf16 v16bf;
typedef __attribute__((ext_vector_type(8)))  float  v8f;
typedef __attribute__((ext_vector_type(4)))  float  f32x4;
typedef __attribute__((ext_vector_type(4)))  unsigned int u32x4;

union FragU { u32x4 u[2]; v16bf v; };

enum { S_ = 1024, D_ = 768, HD_ = 64, H_ = 12, HK_ = 6, KVD_ = 384, FF_ = 3072, VP_ = 50304, NL_ = 8 };
enum { FLAG_ACC = 1, FLAG_SOFTCAP = 2 };

// hardware RNE convert (v_cvt_*bf16_f32) instead of manual bit-twiddling
__device__ __forceinline__ unsigned short f2bf(float f) {
  __bf16 b = (__bf16)f;
  return __builtin_bit_cast(unsigned short, b);
}
__device__ __forceinline__ float bf2f(unsigned short s) {
  return __uint_as_float(((unsigned int)s) << 16);
}

__device__ __forceinline__ v8f vzero8() {
  v8f v;
#pragma unroll
  for (int i = 0; i < 8; ++i) v[i] = 0.f;
  return v;
}

__device__ __forceinline__ v8f wmma_bf16(const FragU& a, const FragU& b, v8f c) {
  return __builtin_amdgcn_wmma_f32_16x16x32_bf16(false, a.v, false, b.v, (short)0, c, false, false);
}

__device__ __forceinline__ float warp_sum32(float v) {
#pragma unroll
  for (int m = 16; m >= 1; m >>= 1) v += __shfl_xor(v, m, 32);
  return v;
}

__device__ __forceinline__ float block_sum256(float v) {
  __shared__ float red[8];
  float s = warp_sum32(v);
  if ((threadIdx.x & 31) == 0) red[threadIdx.x >> 5] = s;
  __syncthreads();
  float t = 0.f;
#pragma unroll
  for (int i = 0; i < 8; ++i) t += red[i];
  __syncthreads();
  return t;
}

// ---------------------------------------------------------------------------
// GEMM: C[1024, N] (+)= A[1024, K](bf16) @ W[N, K](fp32->bf16)^T, bf16 WMMA.
// Block tile 128x64, 8 waves, wave tile 32x32 (2x2 of 16x16), K step 32.
// ---------------------------------------------------------------------------
__global__ __launch_bounds__(256) void gemm_bf16w(
    const unsigned short* __restrict__ A, const float* __restrict__ W,
    float* __restrict__ C, int N, int K, int flags)
{
  __shared__ unsigned short As[128 * 40];
  __shared__ unsigned short Ws[64 * 40];

  const int tid  = threadIdx.x;
  const int lane = tid & 31;
  const int wave = tid >> 5;
  const int wm   = wave >> 1;       // 0..3
  const int wn   = wave & 1;        // 0..1
  const int l15  = lane & 15;
  const int lh   = lane >> 4;
  const int m0   = blockIdx.y * 128;
  const int n0   = blockIdx.x * 64;

  v8f acc[2][2];
#pragma unroll
  for (int i = 0; i < 2; ++i)
#pragma unroll
    for (int j = 0; j < 2; ++j) acc[i][j] = vzero8();

  for (int k0 = 0; k0 < K; k0 += 32) {
    // stage A tile 128x32 bf16: pure 16B copies, 2 chunks/thread
#pragma unroll
    for (int j = 0; j < 2; ++j) {
      const int c    = tid + 256 * j;       // 0..511
      const int row  = c >> 2;              // 0..127
      const int col8 = (c & 3) * 8;         // 0,8,16,24
      const unsigned short* src = A + (size_t)(m0 + row) * K + k0 + col8;
      __builtin_prefetch((const void*)(src + 32), 0, 3);
      *(u32x4*)&As[row * 40 + col8] = *(const u32x4*)src;
    }
    // stage W tile 64x32: fp32 load, hw cvt to bf16
#pragma unroll
    for (int j = 0; j < 2; ++j) {
      const int c   = tid + 256 * j;        // 0..511
      const int row = c >> 3;               // 0..63
      const int col = (c & 7) * 4;
      const float* src = W + (size_t)(n0 + row) * K + k0 + col;
      __builtin_prefetch((const void*)(src + 32), 0, 3);
      const f32x4 v = *(const f32x4*)src;
      unsigned short* d = &Ws[row * 40 + col];
      d[0] = f2bf(v.x); d[1] = f2bf(v.y); d[2] = f2bf(v.z); d[3] = f2bf(v.w);
    }
    __syncthreads();

    FragU a[2], b[2];
#pragma unroll
    for (int mi = 0; mi < 2; ++mi) {
      const unsigned short* p = &As[(wm * 32 + mi * 16 + l15) * 40 + lh * 8];
      a[mi].u[0] = *(const u32x4*)(p + 0);
      a[mi].u[1] = *(const u32x4*)(p + 16);
    }
#pragma unroll
    for (int ni = 0; ni < 2; ++ni) {
      const unsigned short* p = &Ws[(wn * 32 + ni * 16 + l15) * 40 + lh * 8];
      b[ni].u[0] = *(const u32x4*)(p + 0);
      b[ni].u[1] = *(const u32x4*)(p + 16);
    }
#pragma unroll
    for (int mi = 0; mi < 2; ++mi)
#pragma unroll
      for (int ni = 0; ni < 2; ++ni)
        acc[mi][ni] = wmma_bf16(a[mi], b[ni], acc[mi][ni]);
    __syncthreads();
  }

#pragma unroll
  for (int mi = 0; mi < 2; ++mi)
#pragma unroll
    for (int ni = 0; ni < 2; ++ni)
#pragma unroll
      for (int r = 0; r < 8; ++r) {
        const int row = m0 + wm * 32 + mi * 16 + lh * 8 + r;
        const int col = n0 + wn * 32 + ni * 16 + l15;
        const size_t o = (size_t)row * N + col;
        float v = acc[mi][ni][r];
        if (flags & FLAG_ACC)     v += C[o];
        if (flags & FLAG_SOFTCAP) v = 15.f * tanhf(v * (1.f / 15.f));
        C[o] = v;
      }
}

// ---------------------------------------------------------------------------
// Flash attention, WMMA. Grid (H=12, S/128=8), 8 waves x 16 queries each.
// Q/K/V bf16 as [S, nheads, 64]. O computed transposed (O^T = V^T P^T).
// Output Y written as bf16 (feeds the wo GEMM directly).
// ---------------------------------------------------------------------------
__global__ __launch_bounds__(256) void attn_kernel(
    const unsigned short* __restrict__ Q,
    const unsigned short* __restrict__ Kb,
    const unsigned short* __restrict__ Vb,
    unsigned short* __restrict__ Y, int window)
{
  __shared__ unsigned short Pl[8][16 * 32];
  __shared__ unsigned short Vt[8][64 * 40];
  __shared__ float st[8][16];

  const int tid  = threadIdx.x;
  const int lane = tid & 31;
  const int wave = tid >> 5;
  const int l15  = lane & 15;
  const int lh   = lane >> 4;
  const int h    = blockIdx.x;
  const int hk   = h >> 1;                       // GQA repeat=2
  const int q0   = blockIdx.y * 128 + wave * 16;

  FragU qa[2];
  {
    const unsigned short* p = Q + ((size_t)(q0 + l15) * H_ + h) * 64;
#pragma unroll
    for (int j = 0; j < 2; ++j) {
      qa[j].u[0] = *(const u32x4*)(p + j * 32 + lh * 8);
      qa[j].u[1] = *(const u32x4*)(p + j * 32 + 16 + lh * 8);
    }
  }

  v8f o[4];
#pragma unroll
  for (int i = 0; i < 4; ++i) o[i] = vzero8();
  v8f mrow, lrow;
#pragma unroll
  for (int r = 0; r < 8; ++r) { mrow[r] = -1.0e30f; lrow[r] = 0.f; }

  const int kend = q0 + 16;
  int kstart = 0;
  if (window > 0) {
    int lo = q0 - window + 1;
    if (lo < 0) lo = 0;
    kstart = lo & ~31;
  }

  for (int kb0 = kstart; kb0 < kend; kb0 += 32) {
    // stage V chunk transposed: Vt[feat][key], 32 keys x 64 feats
    {
      const unsigned short* vp = Vb + ((size_t)(kb0 + lane) * HK_ + hk) * 64;
#pragma unroll
      for (int c = 0; c < 8; ++c) {
        union { u32x4 u; unsigned short s[8]; } td;
        td.u = *(const u32x4*)(vp + c * 8);
#pragma unroll
        for (int e = 0; e < 8; ++e) Vt[wave][(c * 8 + e) * 40 + lane] = td.s[e];
      }
    }

    // scores for two 16-key chunks
    v8f sc[2];
#pragma unroll
    for (int kc = 0; kc < 2; ++kc) {
      const int kbase = kb0 + kc * 16;
      FragU b0, b1;
      const unsigned short* kp = Kb + ((size_t)(kbase + l15) * HK_ + hk) * 64 + lh * 8;
      b0.u[0] = *(const u32x4*)(kp + 0);
      b0.u[1] = *(const u32x4*)(kp + 16);
      b1.u[0] = *(const u32x4*)(kp + 32);
      b1.u[1] = *(const u32x4*)(kp + 48);
      v8f s = vzero8();
      s = wmma_bf16(qa[0], b0, s);
      s = wmma_bf16(qa[1], b1, s);
      const int kcol = kbase + l15;
#pragma unroll
      for (int r = 0; r < 8; ++r) {
        const int qrow = q0 + lh * 8 + r;
        const bool ok = (kcol <= qrow) && (window <= 0 || kcol > qrow - window);
        s[r] = ok ? s[r] * 0.125f : -3.0e38f;
      }
      sc[kc] = s;
    }

    // row max across chunks + 16 lanes of each half
    v8f cm;
#pragma unroll
    for (int r = 0; r < 8; ++r) cm[r] = fmaxf(sc[0][r], sc[1][r]);
#pragma unroll
    for (int msk = 8; msk >= 1; msk >>= 1)
#pragma unroll
      for (int r = 0; r < 8; ++r) cm[r] = fmaxf(cm[r], __shfl_xor(cm[r], msk, 32));

    v8f mnew, alpha;
#pragma unroll
    for (int r = 0; r < 8; ++r) {
      mnew[r]  = fmaxf(mrow[r], cm[r]);
      alpha[r] = __expf(mrow[r] - mnew[r]);
    }
    v8f p0, p1, rs;
#pragma unroll
    for (int r = 0; r < 8; ++r) {
      p0[r] = __expf(sc[0][r] - mnew[r]);
      p1[r] = __expf(sc[1][r] - mnew[r]);
      rs[r] = p0[r] + p1[r];
    }
#pragma unroll
    for (int msk = 8; msk >= 1; msk >>= 1)
#pragma unroll
      for (int r = 0; r < 8; ++r) rs[r] += __shfl_xor(rs[r], msk, 32);
#pragma unroll
    for (int r = 0; r < 8; ++r) {
      lrow[r] = lrow[r] * alpha[r] + rs[r];
      mrow[r] = mnew[r];
    }

    // bounce P (bf16) + alpha through per-wave LDS
#pragma unroll
    for (int r = 0; r < 8; ++r) {
      Pl[wave][(lh * 8 + r) * 32 + l15]      = f2bf(p0[r]);
      Pl[wave][(lh * 8 + r) * 32 + 16 + l15] = f2bf(p1[r]);
    }
    if (l15 == 0) {
#pragma unroll
      for (int r = 0; r < 8; ++r) st[wave][lh * 8 + r] = alpha[r];
    }
    asm volatile("s_wait_dscnt 0" ::: "memory");   // wave-local LDS ordering

    const float alq = st[wave][l15];               // alpha for query = l15
#pragma unroll
    for (int fs = 0; fs < 4; ++fs)
#pragma unroll
      for (int r = 0; r < 8; ++r) o[fs][r] *= alq;

    FragU pt;                                      // P^T as B fragment
    {
      const unsigned short* p = &Pl[wave][l15 * 32 + lh * 8];
      pt.u[0] = *(const u32x4*)(p + 0);
      pt.u[1] = *(const u32x4*)(p + 16);
    }
#pragma unroll
    for (int fs = 0; fs < 4; ++fs) {               // V^T as A fragment
      FragU av;
      const unsigned short* p = &Vt[wave][(fs * 16 + l15) * 40 + lh * 8];
      av.u[0] = *(const u32x4*)(p + 0);
      av.u[1] = *(const u32x4*)(p + 16);
      o[fs] = wmma_bf16(av, pt, o[fs]);
    }
  }

  if (l15 == 0) {
#pragma unroll
    for (int r = 0; r < 8; ++r) st[wave][lh * 8 + r] = lrow[r];
  }
  asm volatile("s_wait_dscnt 0" ::: "memory");
  const float linv = 1.f / st[wave][l15];
  unsigned short* yp = Y + ((size_t)(q0 + l15) * H_ + h) * 64;
#pragma unroll
  for (int fs = 0; fs < 4; ++fs)
#pragma unroll
    for (int r = 0; r < 8; ++r)
      yp[fs * 16 + lh * 8 + r] = f2bf(o[fs][r] * linv);
}

// ---------------------------------------------------------------------------
// Elementwise / normalization kernels (one block of 256 per token)
// ---------------------------------------------------------------------------
__global__ __launch_bounds__(256) void embed_kernel(
    const int* __restrict__ idx, const float* __restrict__ wte, float* __restrict__ x0)
{
  const int t = blockIdx.x;
  const float* row = wte + (size_t)idx[t] * D_;
  float ss = 0.f;
  for (int d = threadIdx.x; d < D_; d += 256) { const float v = row[d]; ss += v * v; }
  ss = block_sum256(ss);
  const float sc = rsqrtf(ss / (float)D_ + EPSF);
  for (int d = threadIdx.x; d < D_; d += 256) x0[(size_t)t * D_ + d] = row[d] * sc;
}

__global__ __launch_bounds__(256) void smear_kernel(
    const float* __restrict__ x0, const float* __restrict__ sgw,
    const float* __restrict__ sl, float* __restrict__ x)
{
  const int t = blockIdx.x;
  const float* p = x0 + (size_t)t * D_;
  float d = 0.f;
#pragma unroll
  for (int j = 0; j < 24; ++j) d += p[j] * sgw[j];
  const float g = sl[0] / (1.f + __expf(-d));
  for (int dd = threadIdx.x; dd < D_; dd += 256) {
    float v = p[dd];
    if (t > 0) v += g * x0[(size_t)(t - 1) * D_ + dd];
    x[(size_t)t * D_ + dd] = v;
  }
}

__global__ __launch_bounds__(256) void resid_rms_kernel(
    float* __restrict__ x, const float* __restrict__ x0,
    const float* __restrict__ rl, const float* __restrict__ xl, int layer,
    unsigned short* __restrict__ hb)
{
  const int t = blockIdx.x;
  const float a = rl[layer], b = xl[layer];
  float loc[3];
  float ss = 0.f;
#pragma unroll
  for (int i = 0; i < 3; ++i) {
    const int d = threadIdx.x + 256 * i;
    const float v = a * x[(size_t)t * D_ + d] + b * x0[(size_t)t * D_ + d];
    loc[i] = v; ss += v * v;
  }
  ss = block_sum256(ss);
  const float sc = rsqrtf(ss / (float)D_ + EPSF);
#pragma unroll
  for (int i = 0; i < 3; ++i) {
    const int d = threadIdx.x + 256 * i;
    x[(size_t)t * D_ + d]  = loc[i];
    hb[(size_t)t * D_ + d] = f2bf(loc[i] * sc);
  }
}

__global__ __launch_bounds__(256) void rms_kernel(
    const float* __restrict__ x, unsigned short* __restrict__ hb)
{
  const int t = blockIdx.x;
  float loc[3];
  float ss = 0.f;
#pragma unroll
  for (int i = 0; i < 3; ++i) {
    const int d = threadIdx.x + 256 * i;
    const float v = x[(size_t)t * D_ + d];
    loc[i] = v; ss += v * v;
  }
  ss = block_sum256(ss);
  const float sc = rsqrtf(ss / (float)D_ + EPSF);
#pragma unroll
  for (int i = 0; i < 3; ++i)
    hb[(size_t)t * D_ + threadIdx.x + 256 * i] = f2bf(loc[i] * sc);
}

__global__ __launch_bounds__(256) void backout_rms_kernel(
    float* __restrict__ x, const float* __restrict__ x0,
    const float* __restrict__ bl, unsigned short* __restrict__ hb)
{
  const int t = blockIdx.x;
  const float b = bl[0];
  float loc[3];
  float ss = 0.f;
#pragma unroll
  for (int i = 0; i < 3; ++i) {
    const int d = threadIdx.x + 256 * i;
    const float v = x[(size_t)t * D_ + d] - b * x0[(size_t)t * D_ + d];
    loc[i] = v; ss += v * v;
  }
  ss = block_sum256(ss);
  const float sc = rsqrtf(ss / (float)D_ + EPSF);
#pragma unroll
  for (int i = 0; i < 3; ++i) {
    const int d = threadIdx.x + 256 * i;
    x[(size_t)t * D_ + d]  = loc[i];
    hb[(size_t)t * D_ + d] = f2bf(loc[i] * sc);
  }
}

// RoPE + per-head RMS*1.2 for q,k; value-embedding gating for v; bf16 outputs.
__global__ __launch_bounds__(256) void rope_ve_kernel(
    const float* __restrict__ qf, const float* __restrict__ kf,
    const float* __restrict__ vf, const unsigned short* __restrict__ hb,
    const int* __restrict__ idx,
    const float* __restrict__ vet, const float* __restrict__ vgw,
    unsigned short* __restrict__ qb, unsigned short* __restrict__ kb,
    unsigned short* __restrict__ vb, int ve_j)
{
  const int t    = blockIdx.x;
  const int lane = threadIdx.x & 31;
  const int warp = threadIdx.x >> 5;

  // angle for rotary pair j = lane (0..31): inv = 10000^(-j/32)
  const float inv = __expf(-(float)lane * (9.210340372f / 32.f));
  float sn, cs;
  sincosf((float)t * inv, &sn, &cs);

  for (int u = warp; u < 24; u += 8) {
    if (u < 12) {                                    // query heads
      const float* p = qf + (size_t)t * D_ + u * 64;
      const float x1 = p[lane], x2 = p[lane + 32];
      const float r1 =  x1 * cs + x2 * sn;
      const float r2 = -x1 * sn + x2 * cs;
      const float ss = warp_sum32(r1 * r1 + r2 * r2);
      const float sc = rsqrtf(ss / 64.f + EPSF) * 1.2f;
      unsigned short* ob = qb + ((size_t)t * H_ + u) * 64;
      ob[lane] = f2bf(r1 * sc); ob[lane + 32] = f2bf(r2 * sc);
    } else if (u < 18) {                             // key heads
      const int hk = u - 12;
      const float* p = kf + (size_t)t * KVD_ + hk * 64;
      const float x1 = p[lane], x2 = p[lane + 32];
      const float r1 =  x1 * cs + x2 * sn;
      const float r2 = -x1 * sn + x2 * cs;
      const float ss = warp_sum32(r1 * r1 + r2 * r2);
      const float sc = rsqrtf(ss / 64.f + EPSF) * 1.2f;
      unsigned short* ob = kb + ((size_t)t * HK_ + hk) * 64;
      ob[lane] = f2bf(r1 * sc); ob[lane + 32] = f2bf(r2 * sc);
    } else {                                         // value heads
      const int hk = u - 18;
      const float* p = vf + (size_t)t * KVD_ + hk * 64;
      float v1 = p[lane], v2 = p[lane + 32];
      if (ve_j >= 0) {
        const float* gw = vgw + ((size_t)ve_j * HK_ + hk) * 12;
        const unsigned short* hp = hb + (size_t)t * D_;
        float d = 0.f;
#pragma unroll
        for (int j = 0; j < 12; ++j) d += bf2f(hp[j]) * gw[j];
        const float g = 3.f / (1.f + __expf(-d));
        const float* ve = vet + ((size_t)ve_j * VP_ + idx[t]) * KVD_ + hk * 64;
        v1 += g * ve[lane]; v2 += g * ve[lane + 32];
      }
      unsigned short* ob = vb + ((size_t)t * HK_ + hk) * 64;
      ob[lane] = f2bf(v1); ob[lane + 32] = f2bf(v2);
    }
  }
}

// relu(x)^2, fp32 in (GEMM out), bf16 out (next GEMM A operand)
__global__ __launch_bounds__(256) void relu2_kernel(
    const float* __restrict__ f, unsigned short* __restrict__ fb, int n)
{
  const int i = blockIdx.x * 256 + threadIdx.x;
  if (i < n) {
    float v = f[i];
    v = v > 0.f ? v : 0.f;
    fb[i] = f2bf(v * v);
  }
}

// ---------------------------------------------------------------------------
extern "C" void kernel_launch(void* const* d_in, const int* in_sizes, int n_in,
                              void* d_out, int out_size, void* d_ws, size_t ws_size,
                              hipStream_t stream) {
  (void)in_sizes; (void)n_in; (void)out_size; (void)ws_size;
  const int*   idx = (const int*)d_in[0];
  const float* wte = (const float*)d_in[1];
  const float* wq  = (const float*)d_in[2];
  const float* wk  = (const float*)d_in[3];
  const float* wv  = (const float*)d_in[4];
  const float* wo  = (const float*)d_in[5];
  const float* wfc = (const float*)d_in[6];
  const float* wp  = (const float*)d_in[7];
  const float* vet = (const float*)d_in[8];
  const float* vgw = (const float*)d_in[9];
  const float* lmh = (const float*)d_in[10];
  const float* sgw = (const float*)d_in[11];
  const float* rl  = (const float*)d_in[12];
  const float* xl  = (const float*)d_in[13];
  const float* sl  = (const float*)d_in[14];
  const float* bl  = (const float*)d_in[15];
  float* out = (float*)d_out;

  char* wsp = (char*)d_ws;
  auto carve = [&](size_t bytes) -> void* {
    void* p = (void*)wsp;
    wsp += (bytes + 255) & ~(size_t)255;
    return p;
  };
  float* x0 = (float*)carve((size_t)S_ * D_ * 4);
  float* x  = (float*)carve((size_t)S_ * D_ * 4);
  float* qf = (float*)carve((size_t)S_ * D_ * 4);
  float* kf = (float*)carve((size_t)S_ * KVD_ * 4);
  float* vf = (float*)carve((size_t)S_ * KVD_ * 4);
  float* ff = (float*)carve((size_t)S_ * FF_ * 4);
  unsigned short* hb  = (unsigned short*)carve((size_t)S_ * D_ * 2);
  unsigned short* yb  = (unsigned short*)carve((size_t)S_ * D_ * 2);
  unsigned short* ffb = (unsigned short*)carve((size_t)S_ * FF_ * 2);
  unsigned short* qb  = (unsigned short*)carve((size_t)S_ * D_ * 2);
  unsigned short* kb  = (unsigned short*)carve((size_t)S_ * KVD_ * 2);
  unsigned short* vb  = (unsigned short*)carve((size_t)S_ * KVD_ * 2);

  embed_kernel<<<S_, 256, 0, stream>>>(idx, wte, x0);
  smear_kernel<<<S_, 256, 0, stream>>>(x0, sgw, sl, x);

  for (int i = 0; i < NL_; ++i) {
    resid_rms_kernel<<<S_, 256, 0, stream>>>(x, x0, rl, xl, i, hb);

    gemm_bf16w<<<dim3(D_ / 64, S_ / 128), 256, 0, stream>>>(
        hb, wq + (size_t)i * D_ * D_, qf, D_, D_, 0);
    gemm_bf16w<<<dim3(KVD_ / 64, S_ / 128), 256, 0, stream>>>(
        hb, wk + (size_t)i * KVD_ * D_, kf, KVD_, D_, 0);
    gemm_bf16w<<<dim3(KVD_ / 64, S_ / 128), 256, 0, stream>>>(
        hb, wv + (size_t)i * KVD_ * D_, vf, KVD_, D_, 0);

    const int ve_j = (i & 1) ? (i >> 1) : -1;     // VE layers are 1,3,5,7
    rope_ve_kernel<<<S_, 256, 0, stream>>>(qf, kf, vf, hb, idx, vet, vgw,
                                           qb, kb, vb, ve_j);

    const int win = (i & 1) ? -1 : 512;           // WINDOWS = [512,-1,...]
    attn_kernel<<<dim3(H_, S_ / 128), 256, 0, stream>>>(qb, kb, vb, yb, win);

    gemm_bf16w<<<dim3(D_ / 64, S_ / 128), 256, 0, stream>>>(
        yb, wo + (size_t)i * D_ * D_, x, D_, D_, FLAG_ACC);

    rms_kernel<<<S_, 256, 0, stream>>>(x, hb);
    gemm_bf16w<<<dim3(FF_ / 64, S_ / 128), 256, 0, stream>>>(
        hb, wfc + (size_t)i * FF_ * D_, ff, FF_, D_, 0);
    relu2_kernel<<<(S_ * FF_ + 255) / 256, 256, 0, stream>>>(ff, ffb, S_ * FF_);
    gemm_bf16w<<<dim3(D_ / 64, S_ / 128), 256, 0, stream>>>(
        ffb, wp + (size_t)i * D_ * FF_, x, D_, FF_, FLAG_ACC);
  }

  backout_rms_kernel<<<S_, 256, 0, stream>>>(x, x0, bl, hb);
  gemm_bf16w<<<dim3(VP_ / 64, S_ / 128), 256, 0, stream>>>(
      hb, lmh, out, VP_, D_, FLAG_SOFTCAP);
}
